// SACActorNetwork_50775103374015
// MI455X (gfx1250) — matrix operand order repaired and checked by reading, not compiled, not run on hardware
//
#include <hip/hip_runtime.h>
#include <hip/hip_bf16.h>
#include <math.h>

// ---------------------------------------------------------------------------
// SAC actor network forward for MI455X (gfx1250, wave32, WMMA).
//  layer sizes: 256 -> 128 -> 128 -> 64, batch 262144
//  GEMMs on v_wmma_f32_16x16x32_bf16 (bf16 in, f32 accumulate)
//  elementwise softplus / capped-simplex bisection in f32
//  - N-tile loops NOT unrolled (keeps VGPR pressure < 128, no spills)
//  - bisection inner op uses v_med3_f32 with the shift identity
//      clip(w-mid,lo,hi) = med3(w, mid+lo, mid+hi) - mid
//  - softplus via hardware v_exp_f32/v_log_f32 (log1pf libm call was ~50 VALU
//    per element and dominated the static VALU count)
// ---------------------------------------------------------------------------

typedef __attribute__((ext_vector_type(16))) __bf16 v16bf;
typedef __attribute__((ext_vector_type(8)))  __bf16 v8bf;
typedef __attribute__((ext_vector_type(8)))  float  v8f;

#define STATE_DIM 256
#define HIDDEN    128
#define ACT       64
#define HPAD      136   // bf16 elems per LDS activation row (128 + 8 pad -> 4-bank rotation)
#define HEADPAD   68    // f32 elems per LDS head row (64 + 4 pad)
#define WAVES     4     // waves per workgroup
#define BUFBYTES  (16 * HPAD * 2)   // 4352 bytes: 16 rows of HPAD bf16 (== 16*HEADPAD f32)
#define PROJ_ITERS 60

__device__ __forceinline__ float clipf(float x, float lo, float hi) {
  return __builtin_amdgcn_fmed3f(x, lo, hi);   // v_med3_f32 (lo <= hi always here)
}

// stable softplus with hardware transcendentals:
//   softplus(x) = max(x,0) + log(1 + exp(-|x|))
__device__ __forceinline__ float softplus_fast(float x) {
  return fmaxf(x, 0.0f) + __logf(1.0f + __expf(-fabsf(x)));
}

__device__ __forceinline__ v8f wmma_bf16(v16bf a, v16bf b, v8f c) {
  return __builtin_amdgcn_wmma_f32_16x16x32_bf16(false, a, false, b, (short)0, c,
                                                 false, false);
}

__device__ __forceinline__ v16bf cvt_frag(float4 a0, float4 a1, float4 a2, float4 a3) {
  v16bf A;
  A[0]=(__bf16)a0.x;  A[1]=(__bf16)a0.y;  A[2]=(__bf16)a0.z;  A[3]=(__bf16)a0.w;
  A[4]=(__bf16)a1.x;  A[5]=(__bf16)a1.y;  A[6]=(__bf16)a1.z;  A[7]=(__bf16)a1.w;
  A[8]=(__bf16)a2.x;  A[9]=(__bf16)a2.y;  A[10]=(__bf16)a2.z; A[11]=(__bf16)a2.w;
  A[12]=(__bf16)a3.x; A[13]=(__bf16)a3.y; A[14]=(__bf16)a3.z; A[15]=(__bf16)a3.w;
  return A;
}

__device__ __forceinline__ v16bf lds_frag(const __bf16* hp) {
  v8bf lo = *(const v8bf*)hp;
  v8bf hi = *(const v8bf*)(hp + 16);
  v16bf A;
#pragma unroll
  for (int i = 0; i < 8; ++i) { A[i] = lo[i]; A[8 + i] = hi[i]; }
  return A;
}

// ---------------------------------------------------------------------------
// Prep: transpose + f32->bf16 convert the three weight matrices into d_ws.
//   W1T[n][k] = W1[k][n]  (128 x 256)
//   W2T[n][k] = W2[k][n]  (128 x 128)
//   WCT[n][k] = Wc[k][n]  ( 64 x 128)
// ---------------------------------------------------------------------------
__global__ void prep_weights(const float* __restrict__ W1,
                             const float* __restrict__ W2,
                             const float* __restrict__ Wc,
                             __bf16* __restrict__ W1T,
                             __bf16* __restrict__ W2T,
                             __bf16* __restrict__ WCT) {
  int i = blockIdx.x * blockDim.x + threadIdx.x;
  const int N1 = HIDDEN * STATE_DIM;       // 32768
  const int N2 = HIDDEN * HIDDEN;          // 16384
  const int N3 = ACT * HIDDEN;             // 8192
  if (i < N1) {
    int n = i / STATE_DIM, k = i % STATE_DIM;
    W1T[i] = (__bf16)W1[k * HIDDEN + n];
  } else if (i < N1 + N2) {
    int j = i - N1; int n = j / HIDDEN, k = j % HIDDEN;
    W2T[j] = (__bf16)W2[k * HIDDEN + n];
  } else if (i < N1 + N2 + N3) {
    int j = i - (N1 + N2); int n = j / HIDDEN, k = j % HIDDEN;
    WCT[j] = (__bf16)Wc[k * ACT + n];
  }
}

// ---------------------------------------------------------------------------
// Main kernel: one wave per 16-row batch tile.
// ---------------------------------------------------------------------------
__global__ void __launch_bounds__(WAVES * 32) actor_forward(
    const float*  __restrict__ state,
    const float*  __restrict__ b1,
    const float*  __restrict__ b2,
    const float*  __restrict__ bc,
    const __bf16* __restrict__ W1T,
    const __bf16* __restrict__ W2T,
    const __bf16* __restrict__ WCT,
    float* __restrict__ outConc,
    float* __restrict__ outProj,
    float* __restrict__ outLP,
    float* __restrict__ outEnt,
    int batch) {
  // per-wave LDS scratch: two ping-pong buffers.
  //  buf0: layer-1 output (bf16) -> later reused as f32 head tile
  //  buf1: layer-2 output (bf16)
  __shared__ __align__(16) unsigned char smem_raw[WAVES][2][BUFBYTES];

  const int lane = threadIdx.x & 31;
  const int wid  = threadIdx.x >> 5;
  __bf16* buf0  = (__bf16*)smem_raw[wid][0];
  __bf16* buf1  = (__bf16*)smem_raw[wid][1];
  float*  HEADF = (float*)smem_raw[wid][0];

  const int m  = lane & 15;   // A-matrix row / D-matrix column index
  const int kh = lane >> 4;   // K-half selector (low/high lane group)

  const int numTiles = batch >> 4;
  const int tileStep = gridDim.x * WAVES;

  for (int tile = blockIdx.x * WAVES + wid; tile < numTiles; tile += tileStep) {
    const int m0 = tile << 4;

    // hint-prefetch next tile's state rows into cache hierarchy
    {
      int tn = tile + tileStep;
      if (tn < numTiles)
        __builtin_prefetch(state + (size_t)((tn << 4) + m) * STATE_DIM + kh * 128, 0, 0);
    }

    // ---- Layer 1: buf0 = relu(state[m0:m0+16] @ W1 + b1) -----------------
    // A fragments (16x32 bf16 each), loaded once, reused for all 8 N tiles.
    v16bf A1[8];
    const float* srow = state + (size_t)(m0 + m) * STATE_DIM;
#pragma unroll
    for (int kb = 0; kb < 8; ++kb) {
      const float* ap = srow + kb * 32 + kh * 8;
      float4 a0 = *(const float4*)(ap + 0);
      float4 a1 = *(const float4*)(ap + 4);
      float4 a2 = *(const float4*)(ap + 16);
      float4 a3 = *(const float4*)(ap + 20);
      A1[kb] = cvt_frag(a0, a1, a2, a3);
    }
#pragma unroll 1
    for (int nt = 0; nt < 8; ++nt) {
      float bias = b1[nt * 16 + m];
      v8f c;
#pragma unroll
      for (int r = 0; r < 8; ++r) c[r] = bias;
      const __bf16* wrow = W1T + (size_t)(nt * 16 + m) * STATE_DIM + kh * 8;
#pragma unroll
      for (int kb = 0; kb < 8; ++kb) {
        v16bf B = lds_frag(wrow + kb * 32);
        c = wmma_bf16(A1[kb], B, c);
      }
#pragma unroll
      for (int r = 0; r < 8; ++r)
        buf0[(r + 8 * kh) * HPAD + nt * 16 + m] = (__bf16)fmaxf(c[r], 0.0f);
    }

    // ---- Layer 2: buf1 = relu(buf0 @ W2 + b2) ----------------------------
    v16bf A2[4];
#pragma unroll
    for (int kb = 0; kb < 4; ++kb)
      A2[kb] = lds_frag(buf0 + m * HPAD + kb * 32 + kh * 8);
#pragma unroll 1
    for (int nt = 0; nt < 8; ++nt) {
      float bias = b2[nt * 16 + m];
      v8f c;
#pragma unroll
      for (int r = 0; r < 8; ++r) c[r] = bias;
      const __bf16* wrow = W2T + (size_t)(nt * 16 + m) * HIDDEN + kh * 8;
#pragma unroll
      for (int kb = 0; kb < 4; ++kb) {
        v16bf B = lds_frag(wrow + kb * 32);
        c = wmma_bf16(A2[kb], B, c);
      }
#pragma unroll
      for (int r = 0; r < 8; ++r)
        buf1[(r + 8 * kh) * HPAD + nt * 16 + m] = (__bf16)fmaxf(c[r], 0.0f);
    }

    // ---- Layer 3: head = buf1 @ Wc + bc  (f32 into buf0) -----------------
    v16bf A3[4];
#pragma unroll
    for (int kb = 0; kb < 4; ++kb)
      A3[kb] = lds_frag(buf1 + m * HPAD + kb * 32 + kh * 8);
#pragma unroll 1
    for (int nt = 0; nt < 4; ++nt) {
      float bias = bc[nt * 16 + m];
      v8f c;
#pragma unroll
      for (int r = 0; r < 8; ++r) c[r] = bias;
      const __bf16* wrow = WCT + (size_t)(nt * 16 + m) * HIDDEN + kh * 8;
#pragma unroll
      for (int kb = 0; kb < 4; ++kb) {
        v16bf B = lds_frag(wrow + kb * 32);
        c = wmma_bf16(A3[kb], B, c);
      }
#pragma unroll
      for (int r = 0; r < 8; ++r)
        HEADF[(r + 8 * kh) * HEADPAD + nt * 16 + m] = c[r];
    }

    // ---- Elementwise: concentration, weights, bisection projection -------
    // lane pair (p, p+16) splits row p: half kh owns 32 of the 64 actions.
    const int p  = m;
    const int hf = kh;
    const float* hp = HEADF + p * HEADPAD + hf * 32;
    float*       crow = outConc + (size_t)(m0 + p) * ACT + hf * 32;

    float w[32];
    float csum = 0.0f;
#pragma unroll
    for (int j4 = 0; j4 < 8; ++j4) {
      float4 hv = *(const float4*)(hp + j4 * 4);
      float xv[4] = {hv.x, hv.y, hv.z, hv.w};
      float cv[4];
#pragma unroll
      for (int q = 0; q < 4; ++q) {
        float x  = xv[q];
        float hc = clipf(x, -10.0f, 10.0f);
        cv[q] = clipf(softplus_fast(hc) + 1e-4f, 1e-4f, 50.0f);
        float cw = clipf(softplus_fast(x) + 1e-4f, 0.2f, 50.0f);
        w[j4 * 4 + q] = cw;
        csum += cw;
      }
      float4 co = {cv[0], cv[1], cv[2], cv[3]};
      *(float4*)(crow + j4 * 4) = co;
    }
    csum += __shfl_xor(csum, 16, 32);
    float inv = 1.0f / csum;
    float wmn = 1e30f, wmx = -1e30f;
#pragma unroll
    for (int j = 0; j < 32; ++j) {
      w[j] *= inv;
      wmn = fminf(wmn, w[j]);
      wmx = fmaxf(wmx, w[j]);
    }
    wmn = fminf(wmn, __shfl_xor(wmn, 16, 32));
    wmx = fmaxf(wmx, __shfl_xor(wmx, 16, 32));

    const float WMIN = 0.001f;
    const float WMAX = 1.0f - 0.001f * (ACT - 1);
    float lo = wmn - WMAX;
    float hi = wmx - WMIN;
    // bisection on tau; per element: clip(w-mid,WMIN,WMAX) = med3(w,mid+WMIN,mid+WMAX)-mid
    // threshold folded:  sum(med3) > 1 + 64*mid   <=>  sum(clip) > 1
#pragma unroll 1
    for (int it = 0; it < PROJ_ITERS; ++it) {
      float mid = 0.5f * (lo + hi);
      float a = mid + WMIN;
      float b = mid + WMAX;
      float s0 = 0.0f, s1 = 0.0f, s2 = 0.0f, s3 = 0.0f;
#pragma unroll
      for (int j = 0; j < 32; j += 4) {
        s0 += clipf(w[j + 0], a, b);
        s1 += clipf(w[j + 1], a, b);
        s2 += clipf(w[j + 2], a, b);
        s3 += clipf(w[j + 3], a, b);
      }
      float s = (s0 + s1) + (s2 + s3);
      s += __shfl_xor(s, 16, 32);
      float thr = fmaf(64.0f, mid, 1.0f);
      if (s > thr) lo = mid; else hi = mid;
    }
    float tau = 0.5f * (lo + hi);
    float* prow = outProj + (size_t)(m0 + p) * ACT + hf * 32;
#pragma unroll
    for (int j4 = 0; j4 < 8; ++j4) {
      float4 po;
      po.x = clipf(w[j4 * 4 + 0] - tau, WMIN, WMAX);
      po.y = clipf(w[j4 * 4 + 1] - tau, WMIN, WMAX);
      po.z = clipf(w[j4 * 4 + 2] - tau, WMIN, WMAX);
      po.w = clipf(w[j4 * 4 + 3] - tau, WMIN, WMAX);
      *(float4*)(prow + j4 * 4) = po;
    }
    if (hf == 0) outLP[m0 + p] = 0.0f;
    else         outEnt[m0 + p] = 0.0f;

    // compiler-level memory fence: next tile's bf16 LDS stores must not be
    // hoisted above this tile's f32 LDS reads (different TBAA types).
    asm volatile("" ::: "memory");
  }
}

// ---------------------------------------------------------------------------
extern "C" void kernel_launch(void* const* d_in, const int* in_sizes, int n_in,
                              void* d_out, int out_size, void* d_ws, size_t ws_size,
                              hipStream_t stream) {
  (void)n_in; (void)out_size; (void)ws_size;
  const float* state = (const float*)d_in[0];
  const float* W1    = (const float*)d_in[1];
  const float* b1    = (const float*)d_in[2];
  const float* W2    = (const float*)d_in[3];
  const float* b2    = (const float*)d_in[4];
  const float* Wc    = (const float*)d_in[5];
  const float* bc    = (const float*)d_in[6];

  const int batch = in_sizes[0] / STATE_DIM;

  // workspace layout (32B-aligned bf16 weight images): 64KB + 32KB + 16KB
  __bf16* W1T = (__bf16*)d_ws;
  __bf16* W2T = (__bf16*)((char*)d_ws + 65536);
  __bf16* WCT = (__bf16*)((char*)d_ws + 98304);

  float* outConc = (float*)d_out;
  float* outProj = outConc + (size_t)batch * ACT;
  float* outLP   = outProj + (size_t)batch * ACT;
  float* outEnt  = outLP + batch;

  const int prepElems = HIDDEN * STATE_DIM + HIDDEN * HIDDEN + ACT * HIDDEN;
  prep_weights<<<(prepElems + 255) / 256, 256, 0, stream>>>(W1, W2, Wc, W1T, W2T, WCT);

  const int tiles  = batch >> 4;
  int blocks = (tiles + WAVES - 1) / WAVES;
  if (blocks > 8192) blocks = 8192;   // grid-stride covers the rest
  actor_forward<<<blocks, WAVES * 32, 0, stream>>>(state, b1, b2, bc, W1T, W2T, WCT,
                                                   outConc, outProj, outLP, outEnt, batch);
}
